// PVConv_58428735094795
// MI455X (gfx1250) — compile-verified, matrix operand order checked by reading, not run on hardware
//
#include <hip/hip_runtime.h>

#define BATCH 4
#define NPTS  32768
#define VOX   32768
#define RGRID 32
#define CIN   64
#define COUT  128
#define BN_INV 0.99999500003749968f

typedef __bf16 bf16;
typedef __attribute__((ext_vector_type(16))) __bf16 v16bf;
typedef __attribute__((ext_vector_type(8)))  __bf16 v8bf;
typedef __attribute__((ext_vector_type(4)))  __bf16 v4bf;
typedef __attribute__((ext_vector_type(8)))  float  v8f;
typedef int v4i __attribute__((vector_size(16)));

#if defined(__has_builtin)
#  if __has_builtin(__builtin_amdgcn_global_load_async_to_lds_b128)
#    define HAVE_ASYNC 1
#  endif
#endif
#ifndef HAVE_ASYNC
#  define HAVE_ASYNC 0
#endif

// builtin signature (from hipcc diagnostic): src is addrspace(1) int4*, dst LDS int4*
#define GAS(p) ((__attribute__((address_space(1))) v4i*)(p))
#define LAS(p) ((__attribute__((address_space(3))) v4i*)(p))

static __device__ __forceinline__ void wait_async0() {
#if defined(__has_builtin) && __has_builtin(__builtin_amdgcn_s_wait_asynccnt)
    __builtin_amdgcn_s_wait_asynccnt(0);
#else
    asm volatile("s_wait_asynccnt 0" ::: "memory");
#endif
}

// ---------------------------------------------------------------- utilities
__global__ void zero_f32(float* __restrict__ p, int n) {
    int i = blockIdx.x * blockDim.x + threadIdx.x;
    if (i < n) p[i] = 0.0f;
}

// features f32 [B][C][N] -> bf16 [B][N][C]
__global__ void feat_to_nhwc(const float* __restrict__ in, bf16* __restrict__ out) {
    int i = blockIdx.x * blockDim.x + threadIdx.x;
    if (i >= BATCH * NPTS * CIN) return;
    int ch = i % CIN;
    int p  = (i / CIN) % NPTS;
    int b  = i / (CIN * NPTS);
    out[i] = (bf16)in[((size_t)b * CIN + ch) * NPTS + p];
}

// conv weights [O][Cin][T] (f32) -> [O][T][Cin] (bf16)
__global__ void repack_w(const float* __restrict__ w, bf16* __restrict__ out,
                         int O, int Ci, int T) {
    int i = blockIdx.x * blockDim.x + threadIdx.x;
    int n = O * Ci * T;
    if (i >= n) return;
    int t  = i % T;
    int cc = (i / T) % Ci;
    int o  = i / (T * Ci);
    out[((size_t)o * T + t) * Ci + cc] = (bf16)w[i];
}

// fold conv bias into inference BN: y = acc*sc + sh
__global__ void make_scsh(const float* __restrict__ g, const float* __restrict__ beta,
                          const float* __restrict__ cb, float* __restrict__ sc,
                          float* __restrict__ sh, int C) {
    int i = blockIdx.x * blockDim.x + threadIdx.x;
    if (i >= C) return;
    float s = g[i] * BN_INV;
    sc[i] = s;
    sh[i] = cb[i] * s + beta[i];
}

// ---------------------------------------------------------------- voxelize
__launch_bounds__(256)
__global__ void vox_stats(const float* __restrict__ coords, float* __restrict__ ms) {
    __shared__ float red[256];
    const int b = blockIdx.x;
    const float* cx = coords + (size_t)b * 3 * NPTS;
    float s3[3] = {0.f, 0.f, 0.f};
    for (int i = threadIdx.x; i < NPTS; i += 256) {
        s3[0] += cx[i]; s3[1] += cx[NPTS + i]; s3[2] += cx[2 * NPTS + i];
    }
    float m[3];
    for (int d = 0; d < 3; ++d) {
        red[threadIdx.x] = s3[d];
        __syncthreads();
        for (int o = 128; o > 0; o >>= 1) {
            if (threadIdx.x < o) red[threadIdx.x] += red[threadIdx.x + o];
            __syncthreads();
        }
        m[d] = red[0] * (1.0f / NPTS);
        __syncthreads();
    }
    float mx = 0.f;
    for (int i = threadIdx.x; i < NPTS; i += 256) {
        float dx = cx[i] - m[0], dy = cx[NPTS + i] - m[1], dz = cx[2 * NPTS + i] - m[2];
        mx = fmaxf(mx, sqrtf(dx * dx + dy * dy + dz * dz));
    }
    red[threadIdx.x] = mx;
    __syncthreads();
    for (int o = 128; o > 0; o >>= 1) {
        if (threadIdx.x < o) red[threadIdx.x] = fmaxf(red[threadIdx.x], red[threadIdx.x + o]);
        __syncthreads();
    }
    if (threadIdx.x == 0) {
        ms[b * 4 + 0] = m[0]; ms[b * 4 + 1] = m[1]; ms[b * 4 + 2] = m[2];
        ms[b * 4 + 3] = red[0] * 2.0f;
    }
}

__global__ void vox_scatter(const float* __restrict__ coords, const float* __restrict__ feats,
                            const int* __restrict__ targ, const float* __restrict__ ms,
                            float* __restrict__ nc, float* __restrict__ sums,
                            float* __restrict__ tsum, float* __restrict__ cnt) {
    int gid = blockIdx.x * blockDim.x + threadIdx.x;
    if (gid >= BATCH * NPTS) return;
    int b = gid / NPTS, p = gid - b * NPTS;
    float inv = 1.0f / ms[b * 4 + 3];
    int vi[3];
    for (int d = 0; d < 3; ++d) {
        float c = coords[((size_t)b * 3 + d) * NPTS + p];
        float v = ((c - ms[b * 4 + d]) * inv + 0.5f) * (float)RGRID;
        v = fminf(fmaxf(v, 0.0f), (float)(RGRID - 1));
        nc[((size_t)b * 3 + d) * NPTS + p] = v;
        vi[d] = __float2int_rn(v);
    }
    int idx = vi[0] * 1024 + vi[1] * 32 + vi[2];
    atomicAdd(&cnt[(size_t)b * VOX + idx], 1.0f);
    atomicAdd(&tsum[(size_t)b * VOX + idx], (float)targ[gid]);
    for (int c = 0; c < CIN; ++c)
        atomicAdd(&sums[((size_t)b * CIN + c) * VOX + idx],
                  feats[((size_t)b * CIN + c) * NPTS + p]);
}

// sums f32 [B][C][V] -> vf bf16 [B][V][C] ; vt = tsum/cnt
__global__ void vox_finalize(const float* __restrict__ sums, const float* __restrict__ tsum,
                             const float* __restrict__ cnt, bf16* __restrict__ vf,
                             float* __restrict__ vt) {
    int gid = blockIdx.x * blockDim.x + threadIdx.x;
    if (gid >= BATCH * VOX) return;
    int b = gid / VOX, v = gid - b * VOX;
    float ic = 1.0f / fmaxf(cnt[gid], 1.0f);
    for (int ch = 0; ch < CIN; ++ch)
        vf[((size_t)b * VOX + v) * CIN + ch] =
            (bf16)(sums[((size_t)b * CIN + ch) * VOX + v] * ic);
    vt[gid] = tsum[gid] * ic;
}

// -------------------------------------------------- implicit-GEMM conv (WMMA)
// Activations are [V][C] bf16 (channel-contiguous). One block = 16-voxel
// W-strip; wave w computes cout tile [16w,16w+16). LDS holds the
// (3x3x18 | 16) x CI halo tile, filled with async global->LDS b128 copies.
template <int CI, int K3>
__launch_bounds__(256)
__global__ void conv_wmma(const bf16* __restrict__ in, const bf16* __restrict__ wq,
                          const float* __restrict__ sc, const float* __restrict__ sh,
                          const bf16* __restrict__ res, bf16* __restrict__ out,
                          int Co, int relu) {
    constexpr int NPOS = K3 ? 9 * 18 : 16;
    constexpr int T    = K3 ? 27 : 1;
    __shared__ bf16 tile[NPOS * CI];

    const int b = blockIdx.z;
    const int s = blockIdx.x;
    const bf16* inb = in + (size_t)b * VOX * CI;
    const int d0 = s >> 6, h0 = (s >> 1) & 31, w0 = (s & 1) << 4;

    // ---- stage input tile: NPOS rows of CI contiguous bf16 each ----
    constexpr int C8 = CI / 8;                 // 16-byte chunks per row
    for (int idx = threadIdx.x; idx < NPOS * C8; idx += blockDim.x) {
        const int pos = idx / C8;
        const int c8  = idx - pos * C8;
        bf16* ldst = tile + pos * CI + c8 * 8;
        int vox; bool oob = false;
        if (K3) {
            const int ww = (pos % 18) - 1 + w0;
            const int hh = h0 + ((pos / 18) % 3) - 1;
            const int dd = d0 + (pos / 54) - 1;
            oob = (dd < 0 || dd >= RGRID || hh < 0 || hh >= RGRID || ww < 0 || ww >= RGRID);
            vox = (dd * RGRID + hh) * RGRID + ww;
        } else {
            vox = s * 16 + pos;
        }
        if (oob) {
            v8bf z = {};
            *(v8bf*)ldst = z;
        } else {
#if HAVE_ASYNC
            __builtin_amdgcn_global_load_async_to_lds_b128(
                GAS((void*)(inb + (size_t)vox * CI + c8 * 8)), LAS(ldst), 0, 0);
#else
            *(v8bf*)ldst = *(const v8bf*)(inb + (size_t)vox * CI + c8 * 8);
#endif
        }
    }
#if HAVE_ASYNC
    wait_async0();
#endif
    __syncthreads();

    const int lane  = threadIdx.x & 31;
    const int wave  = threadIdx.x >> 5;
    const int obase = wave * 16;
    if (obase >= Co) return;
    const int m    = lane & 15;   // A row / B col / D col
    const int half = lane >> 4;   // K-half selector

    const bf16* wbase = wq + (size_t)(obase + m) * T * CI;
    __builtin_prefetch(wbase, 0, 0);   // warm WGP$/L2 with this wave's weight rows

    v8f acc = {};
#pragma unroll
    for (int t = 0; t < T; ++t) {
        int posBase = 0;
        if (K3) {
            const int kd = t / 9, kh = (t / 3) % 3, kw = t - kd * 9 - kh * 3;
            posBase = (kd * 3 + kh) * 18 + kw;   // + n = neighbor slot for voxel n
        }
        const bf16* wrow = wbase + t * CI;
        const bf16* brow = tile + (posBase + m) * CI;
#pragma unroll
        for (int c = 0; c < CI; c += 32) {
            union { v16bf v; v8bf h[2]; } A, Bm;
            // 16-bit frag layout: elem i -> K = (i<8?0:16) + 8*half + (i&7)
            A.h[0]  = *(const v8bf*)(wrow + c + half * 8);
            A.h[1]  = *(const v8bf*)(wrow + c + 16 + half * 8);
            Bm.h[0] = *(const v8bf*)(brow + c + half * 8);
            Bm.h[1] = *(const v8bf*)(brow + c + 16 + half * 8);
            acc = __builtin_amdgcn_wmma_f32_16x16x32_bf16(
                false, A.v, false, Bm.v, (short)0, acc, false, false);
        }
    }

    // ---- epilogue: D lane holds voxel (s*16+m), channels obase+8*half+0..7 ----
    const size_t orow = ((size_t)b * VOX + (s * 16 + m)) * Co + obase + half * 8;
    union { v8bf v; bf16 e[8]; } R;
    if (res) {
        R.v = *(const v8bf*)(res + orow);
    } else {
        v8bf z = {};
        R.v = z;
    }
    union { v8bf v; bf16 e[8]; } O;
#pragma unroll
    for (int j = 0; j < 8; ++j) {
        const int o = obase + half * 8 + j;
        float val = acc[j] * sc[o] + sh[o] + (res ? (float)R.e[j] : 0.0f);
        if (relu) val = fmaxf(val, 0.0f);
        O.e[j] = (bf16)val;
    }
    *(v8bf*)(out + orow) = O.v;
}

// -------------------------------------------- trilinear devox + attention fuse
// xg: [B][V][128] bf16, pf: [B][N][128] bf16, out fused: [B][128][N] f32
__launch_bounds__(256)
__global__ void fuse_attn(const bf16* __restrict__ xg, const bf16* __restrict__ pf,
                          const float* __restrict__ nc,
                          const float* __restrict__ aw, const float* __restrict__ ab,
                          const float* __restrict__ bw, const float* __restrict__ bb,
                          const float* __restrict__ qw, const float* __restrict__ qb,
                          float* __restrict__ outf) {
    __shared__ float svp[8][COUT];
    __shared__ float spf[8][COUT];
    const int wave = threadIdx.x >> 5, lane = threadIdx.x & 31;
    const int pid = blockIdx.x * 8 + wave;
    const int b = pid / NPTS, p = pid - b * NPTS;

    const float n0 = nc[((size_t)b * 3 + 0) * NPTS + p];
    const float n1 = nc[((size_t)b * 3 + 1) * NPTS + p];
    const float n2 = nc[((size_t)b * 3 + 2) * NPTS + p];
    const int l0 = (int)floorf(n0), l1 = (int)floorf(n1), l2 = (int)floorf(n2);
    const float f0 = n0 - l0, f1 = n1 - l1, f2 = n2 - l2;
    const int u0 = min(l0 + 1, 31), u1 = min(l1 + 1, 31), u2 = min(l2 + 1, 31);

    const bf16* gb = xg + (size_t)b * VOX * COUT;
    float a4[4] = {0.f, 0.f, 0.f, 0.f};
#pragma unroll
    for (int k = 0; k < 8; ++k) {
        const int dx = k >> 2, dy = (k >> 1) & 1, dz = k & 1;
        const int ci = (dx ? u0 : l0) * 1024 + (dy ? u1 : l1) * 32 + (dz ? u2 : l2);
        const float w = (dx ? f0 : 1.f - f0) * (dy ? f1 : 1.f - f1) * (dz ? f2 : 1.f - f2);
        v4bf tv = *(const v4bf*)(gb + (size_t)ci * COUT + lane * 4);
#pragma unroll
        for (int j = 0; j < 4; ++j) a4[j] += w * (float)tv[j];
    }
    v4bf pv = *(const v4bf*)(pf + ((size_t)b * NPTS + p) * COUT + lane * 4);
#pragma unroll
    for (int j = 0; j < 4; ++j) {
        svp[wave][lane * 4 + j] = a4[j];
        spf[wave][lane * 4 + j] = (float)pv[j];
    }
    __syncthreads();
    float xa = ab[lane], xb = bb[lane];
    for (int c = 0; c < COUT; ++c) {
        xa += aw[lane * COUT + c] * svp[wave][c];
        xb += bw[lane * COUT + c] * spf[wave][c];
    }
    float qp = qw[lane] * xa + qw[32 + lane] * xb;
    for (int o = 16; o > 0; o >>= 1) qp += __shfl_xor(qp, o, 32);
    const float q   = qp + qb[0];
    const float att = 1.0f + 1.0f / (1.0f + expf(-q));
#pragma unroll
    for (int j = 0; j < 4; ++j) {
        const int ch = lane * 4 + j;
        outf[((size_t)b * COUT + ch) * NPTS + p] = svp[wave][ch] * att + spf[wave][ch];
    }
}

__global__ void vout_head(const bf16* __restrict__ xg, const float* __restrict__ ow,
                          const float* __restrict__ ob, float* __restrict__ vout) {
    int gid = blockIdx.x * blockDim.x + threadIdx.x;
    if (gid >= BATCH * VOX) return;
    const bf16* row = xg + (size_t)gid * COUT;
    float s = ob[0];
    for (int o = 0; o < COUT; ++o) s += ow[o] * (float)row[o];
    vout[gid] = s;
}

// ---------------------------------------------------------------- host glue
static inline int cdiv(int a, int b) { return (a + b - 1) / b; }

extern "C" void kernel_launch(void* const* d_in, const int* in_sizes, int n_in,
                              void* d_out, int out_size, void* d_ws, size_t ws_size,
                              hipStream_t stream) {
    const float* features = (const float*)d_in[0];
    const float* coords   = (const float*)d_in[1];
    const int*   targets  = (const int*)d_in[2];
    const float* P[36];
    for (int i = 0; i < 36; ++i) P[i] = (const float*)d_in[3 + i];
    // 0 c0_w 1 c0_b 2 bn0_g 3 bn0_b | 4-7 r1a | 8-11 r1b | 12-15 r1d | 16-19 r2a
    // 20-23 r2b | 24 out_w 25 out_b | 26-29 mlp | 30-33 a/b | 34 q_w 35 q_b

    char* ws = (char*)d_ws;
    const size_t off_nc   = 256;
    const size_t off_cnt  = off_nc + (size_t)BATCH * 3 * NPTS * 4;
    const size_t off_tsum = off_cnt + (size_t)BATCH * VOX * 4;
    const size_t off_scsh = off_tsum + (size_t)BATCH * VOX * 4;
    const size_t off_wbuf = off_scsh + 7 * 256 * 4;
    const size_t wo_c0  = 0;
    const size_t wo_r1a = wo_c0  + (size_t)64  * 27 * 64;
    const size_t wo_r1b = wo_r1a + (size_t)128 * 27 * 64;
    const size_t wo_r1d = wo_r1b + (size_t)128 * 27 * 128;
    const size_t wo_r2a = wo_r1d + (size_t)128 * 1  * 64;
    const size_t wo_r2b = wo_r2a + (size_t)128 * 27 * 128;
    const size_t wo_mlp = wo_r2b + (size_t)128 * 27 * 128;
    const size_t wbuf_elems = wo_mlp + (size_t)128 * 1 * 64;
    size_t big = off_wbuf + wbuf_elems * 2;
    big = (big + 255) & ~(size_t)255;
    const size_t SLOT = (size_t)BATCH * COUT * VOX * 2;   // 33.55 MB (== f32 64ch)
    const size_t slotA = big, slotB = slotA + SLOT, slotC = slotB + SLOT;
    const size_t slotD = slotC + SLOT;                    // 16.78 MB (64ch bf16)

    float* ms    = (float*)(ws + 0);
    float* ncbuf = (float*)(ws + off_nc);
    float* cnt   = (float*)(ws + off_cnt);
    float* tsum  = (float*)(ws + off_tsum);
    float* scsh  = (float*)(ws + off_scsh);
    bf16*  wbuf  = (bf16*)(ws + off_wbuf);
    float* sums  = (float*)(ws + slotA);    // f32 [B][64][V] scatter accumulator
    bf16*  bufA  = (bf16*)(ws + slotA);
    bf16*  bufB  = (bf16*)(ws + slotB);
    bf16*  bufC  = (bf16*)(ws + slotC);
    bf16*  bufD  = (bf16*)(ws + slotD);

    float* out_fused  = (float*)d_out;                           // [B,128,N]
    float* out_coords = out_fused + (size_t)BATCH * COUT * NPTS; // [B,3,N]
    float* out_vout   = out_coords + (size_t)BATCH * 3 * NPTS;   // [B,1,32^3]
    float* out_vt     = out_vout + (size_t)BATCH * VOX;          // [B,1,32^3]

    // ---- weight repack + BN folding ----
    struct { const float* w; size_t off; int O, Ci, T; } WR[7] = {
        {P[0],  wo_c0,  64, 64, 27},  {P[4],  wo_r1a, 128, 64, 27},
        {P[8],  wo_r1b, 128, 128, 27}, {P[12], wo_r1d, 128, 64, 1},
        {P[16], wo_r2a, 128, 128, 27}, {P[20], wo_r2b, 128, 128, 27},
        {P[26], wo_mlp, 128, 64, 1}};
    for (int i = 0; i < 7; ++i) {
        int n = WR[i].O * WR[i].Ci * WR[i].T;
        repack_w<<<cdiv(n, 256), 256, 0, stream>>>(WR[i].w, wbuf + WR[i].off,
                                                   WR[i].O, WR[i].Ci, WR[i].T);
    }
    struct { const float* g; const float* bt; const float* cb; int C; } SC[7] = {
        {P[2], P[3], P[1], 64},    {P[6], P[7], P[5], 128},   {P[10], P[11], P[9], 128},
        {P[14], P[15], P[13], 128}, {P[18], P[19], P[17], 128}, {P[22], P[23], P[21], 128},
        {P[28], P[29], P[27], 128}};
    for (int i = 0; i < 7; ++i)
        make_scsh<<<1, 128, 0, stream>>>(SC[i].g, SC[i].bt, SC[i].cb,
                                         scsh + i * 256, scsh + i * 256 + 128, SC[i].C);
#define SCL(i) (scsh + (i) * 256)
#define SHF(i) (scsh + (i) * 256 + 128)

    // ---- voxelization ----
    zero_f32<<<cdiv(BATCH * CIN * VOX, 256), 256, 0, stream>>>(sums, BATCH * CIN * VOX);
    zero_f32<<<cdiv(BATCH * VOX, 256), 256, 0, stream>>>(cnt, BATCH * VOX);
    zero_f32<<<cdiv(BATCH * VOX, 256), 256, 0, stream>>>(tsum, BATCH * VOX);
    vox_stats<<<BATCH, 256, 0, stream>>>(coords, ms);
    vox_scatter<<<cdiv(BATCH * NPTS, 256), 256, 0, stream>>>(
        coords, features, targets, ms, ncbuf, sums, tsum, cnt);
    vox_finalize<<<cdiv(BATCH * VOX, 256), 256, 0, stream>>>(sums, tsum, cnt, bufD, out_vt);

    // ---- conv pipeline (WMMA implicit GEMM), activations [V][C] bf16 ----
    dim3 grid(VOX / 16, 1, BATCH);
    // conv0: 64->64, vf(D) -> x0(C)
    conv_wmma<64, 1><<<grid, 128, 0, stream>>>(bufD, wbuf + wo_c0, SCL(0), SHF(0),
                                               nullptr, bufC, 64, 1);
    // r1a: 64->128, x0(C) -> y1(A)
    conv_wmma<64, 1><<<grid, 256, 0, stream>>>(bufC, wbuf + wo_r1a, SCL(1), SHF(1),
                                               nullptr, bufA, 128, 1);
    // r1d: 1x1 64->128, x0(C) -> d(B)
    conv_wmma<64, 0><<<grid, 256, 0, stream>>>(bufC, wbuf + wo_r1d, SCL(3), SHF(3),
                                               nullptr, bufB, 128, 0);
    // r1b: 128->128 + d, y1(A) -> x1(C)
    conv_wmma<128, 1><<<grid, 256, 0, stream>>>(bufA, wbuf + wo_r1b, SCL(2), SHF(2),
                                                bufB, bufC, 128, 1);
    // r2a: 128->128, x1(C) -> y2(A)
    conv_wmma<128, 1><<<grid, 256, 0, stream>>>(bufC, wbuf + wo_r2a, SCL(4), SHF(4),
                                                nullptr, bufA, 128, 1);
    // r2b: 128->128 + x1, y2(A) -> x2(B)
    conv_wmma<128, 1><<<grid, 256, 0, stream>>>(bufA, wbuf + wo_r2b, SCL(5), SHF(5),
                                                bufC, bufB, 128, 1);

    // ---- point branch: features -> [N][64] bf16 (D), mlp via 1x1 WMMA -> pf(A) ----
    feat_to_nhwc<<<cdiv(BATCH * CIN * NPTS, 256), 256, 0, stream>>>(features, bufD);
    conv_wmma<64, 0><<<grid, 256, 0, stream>>>(bufD, wbuf + wo_mlp, SCL(6), SHF(6),
                                               nullptr, bufA, 128, 1);

    // ---- devox + attention fusion, voxel head, coords passthrough ----
    fuse_attn<<<BATCH * NPTS / 8, 256, 0, stream>>>(bufB, bufA, ncbuf,
        P[30], P[31], P[32], P[33], P[34], P[35], out_fused);
    vout_head<<<cdiv(BATCH * VOX, 256), 256, 0, stream>>>(bufB, P[24], P[25], out_vout);
    (void)hipMemcpyAsync(out_coords, coords, (size_t)BATCH * 3 * NPTS * sizeof(float),
                         hipMemcpyDeviceToDevice, stream);
}
